// SDMessengerTransformer_79362405695996
// MI455X (gfx1250) — matrix operand order repaired and checked by reading, not compiled, not run on hardware
//
#include <hip/hip_runtime.h>
#include <math.h>

// ======================= types for CDNA5 WMMA =======================
typedef __bf16 bf16_t;
typedef bf16_t v2bf  __attribute__((ext_vector_type(2)));
typedef bf16_t v8bf  __attribute__((ext_vector_type(8)));
typedef bf16_t v16bf __attribute__((ext_vector_type(16)));
typedef float  v8f   __attribute__((ext_vector_type(8)));

// ======================= small device helpers =======================
__device__ __forceinline__ float gelu_exact(float x) {
  return 0.5f * x * (1.0f + erff(x * 0.70710678118654752440f));
}

__device__ __forceinline__ float block_sum(float v, float* sb) {
  int t = threadIdx.x;
  sb[t] = v; __syncthreads();
  for (int s = blockDim.x >> 1; s > 0; s >>= 1) {
    if (t < s) sb[t] += sb[t + s];
    __syncthreads();
  }
  float r = sb[0]; __syncthreads();
  return r;
}

__device__ __forceinline__ float block_max(float v, float* sb) {
  int t = threadIdx.x;
  sb[t] = v; __syncthreads();
  for (int s = blockDim.x >> 1; s > 0; s >>= 1) {
    if (t < s) sb[t] = fmaxf(sb[t], sb[t + s]);
    __syncthreads();
  }
  float r = sb[0]; __syncthreads();
  return r;
}

// jax.image.resize 'bilinear': half-pixel centers, edge clamp
__device__ __forceinline__ float sample_bilinear(const float* plane, int S, float yin, float xin) {
  float y = fminf(fmaxf(yin, 0.f), (float)(S - 1));
  float x = fminf(fmaxf(xin, 0.f), (float)(S - 1));
  int y0 = (int)floorf(y), x0 = (int)floorf(x);
  int y1 = y0 + 1 < S ? y0 + 1 : S - 1;
  int x1 = x0 + 1 < S ? x0 + 1 : S - 1;
  float fy = y - (float)y0, fx = x - (float)x0;
  float a = plane[y0 * S + x0], b = plane[y0 * S + x1];
  float c = plane[y1 * S + x0], d = plane[y1 * S + x1];
  return a * (1.f - fy) * (1.f - fx) + b * (1.f - fy) * fx +
         c * fy * (1.f - fx) + d * fy * fx;
}

// ======================= generic strided WMMA GEMM =======================
// C[z][m][n] = act( sum_k A[z][m][k] * B[z][k][n] + bias[n] )
// Strides in ELEMENTS.  ABF/BBF/CBF: compile-time "element is bf16" flags.
// GUARD: compile-time bounds-check flag.  GUARD=0 requires M%64==N%64==0 and
// K%32==0 and gives fully branchless, speculation-friendly staging loads.
//
// Block = 4 wave32's.  Block tile 64(M) x 64(N); each wave owns 16 M-rows and
// all 64 N-cols via 4 f32 accumulators -> 4 back-to-back v_wmma per K-step,
// A-fragment reused across the 4 B-fragments.
//
// LDS is laid out so every fragment is loadable with ds_load_b128:
//   As   [m][k]  64x32 (rows padded to 40 halves = 80B, 16B-aligned rows):
//        lane fragment = 2 contiguous 8-half runs (K=8*hi8+0..7, 16+8*hi8+0..7)
//   Bs_t [n][k]  64x32 transposed tile (same padding):
//        lane fragment = 2 contiguous 8-half runs (K=8*hi8+0..7, 16+8*hi8+0..7 at fixed n)
// Fragments per ISA 7.12.2 wave32 layouts:
//   A 16x32 bf16: lane L (m=L&15): half h -> K = 16*(h>>3) + 8*(L>>4) + (h&7)
//   B 32x16 bf16: lane L (n=L&15): half h -> K = 16*(h>>3) + 8*(L>>4) + (h&7)
//   C/D f32:      VGPR r -> m = 8*(L>>4) + r, n = L&15
template <int ABF, int BBF, int CBF, int GUARD>
__global__ __launch_bounds__(128)
void gemm_wmma_t(const void* __restrict__ A, long long a_rs, long long a_cs, long long a_bs,
                 const void* __restrict__ Bm, long long b_rs, long long b_cs, long long b_bs,
                 void* __restrict__ Cm, long long c_rs, long long c_cs, long long c_bs,
                 const float* __restrict__ bias, int act,
                 int M, int N, int K)
{
  __shared__ __align__(16) bf16_t As[64][40];    // 64x32 tile, 80B row stride
  __shared__ __align__(16) bf16_t Bs_t[64][40];  // 32x64 tile stored [n][k]

  const int lane = threadIdx.x;
  const int wave = threadIdx.y;
  const int t    = wave * 32 + lane;
  const int n0   = blockIdx.x * 64;
  const int m0   = blockIdx.y * 64;
  const int bz   = blockIdx.z;

  v8f acc[4];
  #pragma unroll
  for (int j = 0; j < 4; ++j)
    #pragma unroll
    for (int i = 0; i < 8; ++i) acc[j][i] = 0.f;

  const int mrow = wave * 16 + (lane & 15);
  const int hi8  = (lane >> 4) & 1;
  const int ncol = lane & 15;
  const int Mm1 = M - 1, Nm1 = N - 1, Km1 = K - 1;
  const long long zA = (long long)bz * a_bs;
  const long long zB = (long long)bz * b_bs;

  for (int k0 = 0; k0 < K; k0 += 32) {
    // ---- stage A 64x32 tile: 2 elems/thread/iter, packed 4B stores ----
    #pragma unroll
    for (int i = 0; i < 8; ++i) {
      int idx = t + i * 128;           // pair index 0..1023
      int mm = idx >> 4, kk = (idx & 15) << 1;
      int gm = m0 + mm;
      int cm = (GUARD && gm >= M) ? Mm1 : gm;
      v2bf pr;
      #pragma unroll
      for (int e = 0; e < 2; ++e) {
        int gk = k0 + kk + e;
        int ck = (GUARD && gk >= K) ? Km1 : gk;
        long long off = (long long)cm * a_rs + (long long)ck * a_cs + zA;
        float v = ABF ? (float)((const bf16_t*)A)[off] : ((const float*)A)[off];
        if (GUARD) v = (gm < M && gk < K) ? v : 0.f;
        pr[e] = (bf16_t)v;
      }
      *(v2bf*)&As[mm][kk] = pr;
    }
    // ---- stage B 32x64 tile, transposed into Bs_t[n][k] ----
    #pragma unroll
    for (int i = 0; i < 8; ++i) {
      int idx = t + i * 128;
      int nn = idx >> 4, kk = (idx & 15) << 1;
      int gn = n0 + nn;
      int cn = (GUARD && gn >= N) ? Nm1 : gn;
      v2bf pr;
      #pragma unroll
      for (int e = 0; e < 2; ++e) {
        int gk = k0 + kk + e;
        int ck = (GUARD && gk >= K) ? Km1 : gk;
        long long off = (long long)ck * b_rs + (long long)cn * b_cs + zB;
        float v = BBF ? (float)((const bf16_t*)Bm)[off] : ((const float*)Bm)[off];
        if (GUARD) v = (gn < N && gk < K) ? v : 0.f;
        pr[e] = (bf16_t)v;
      }
      *(v2bf*)&Bs_t[nn][kk] = pr;
    }
    // prefetch next K-slab of A (lowers to global_prefetch_b8)
    if (k0 + 32 < K) {
      long long poff = (long long)(m0 + (t >> 1)) * a_rs + (long long)(k0 + 32) * a_cs + zA;
      __builtin_prefetch((const char*)A + poff * (ABF ? 2 : 4), 0, 1);
    }
    __syncthreads();

    // ---- fragments via ds_load_b128 + 4 WMMAs (A fragment reused) ----
    v8bf a0 = *(const v8bf*)&As[mrow][hi8 << 3];
    v8bf a1 = *(const v8bf*)&As[mrow][16 + (hi8 << 3)];
    v16bf af = __builtin_shufflevector(a0, a1, 0, 1, 2, 3, 4, 5, 6, 7,
                                       8, 9, 10, 11, 12, 13, 14, 15);
    #pragma unroll
    for (int j = 0; j < 4; ++j) {
      const bf16_t* brow = &Bs_t[(j << 4) + ncol][hi8 << 3];
      v8bf b0 = *(const v8bf*)(brow);
      v8bf b1 = *(const v8bf*)(brow + 16);
      v16bf bfv = __builtin_shufflevector(b0, b1, 0, 1, 2, 3, 4, 5, 6, 7,
                                          8, 9, 10, 11, 12, 13, 14, 15);
      acc[j] = __builtin_amdgcn_wmma_f32_16x16x32_bf16(false, af, false, bfv,
                                                       (short)0, acc[j], false, false);
    }
    __syncthreads();
  }

  // ---- epilogue ----
  #pragma unroll
  for (int j = 0; j < 4; ++j) {
    int gn = n0 + (j << 4) + ncol;
    #pragma unroll
    for (int r = 0; r < 8; ++r) {
      int gm = m0 + wave * 16 + hi8 * 8 + r;
      if (!GUARD || (gm < M && gn < N)) {
        float v = acc[j][r];
        if (bias) v += bias[gn];
        if (act == 1) v = gelu_exact(v);
        long long co = (long long)gm * c_rs + (long long)gn * c_cs + (long long)bz * c_bs;
        if (CBF) ((bf16_t*)Cm)[co] = (bf16_t)v;
        else     ((float*)Cm)[co]  = v;
      }
    }
  }
}

// ======================= auxiliary kernels =======================
__global__ void zero_kernel(float* p, int n) {
  int i = blockIdx.x * blockDim.x + threadIdx.x;
  if (i < n) p[i] = 0.f;
}

// row-wise layernorm, optional residual add; x/out are [rows][cols] contiguous
__global__ void ln_rows_kernel(const float* __restrict__ x, const float* __restrict__ res,
                               const float* __restrict__ g, const float* __restrict__ b,
                               float* __restrict__ out, int cols, float eps)
{
  __shared__ float sb[256];
  long long base = (long long)blockIdx.x * cols;
  float s = 0.f, s2 = 0.f;
  for (int c = threadIdx.x; c < cols; c += blockDim.x) {
    float v = x[base + c];
    if (res) v += res[base + c];
    s += v; s2 += v * v;
  }
  s  = block_sum(s, sb);
  s2 = block_sum(s2, sb);
  float mean = s / (float)cols;
  float var  = s2 / (float)cols - mean * mean;
  float rstd = rsqrtf(var + eps);
  for (int c = threadIdx.x; c < cols; c += blockDim.x) {
    float v = x[base + c];
    if (res) v += res[base + c];
    out[base + c] = (v - mean) * rstd * g[c] + b[c];
  }
}

// sum / sumsq over contiguous groups (group g = blockIdx.z), atomics into stats[g*2..]
__global__ void reduce_sums_kernel(const float* __restrict__ x, long long group_size,
                                   float* __restrict__ stats)
{
  __shared__ float sb[256];
  int g = blockIdx.z;
  const float* xg = x + (long long)g * group_size;
  float s = 0.f, s2 = 0.f;
  long long stride = (long long)gridDim.x * blockDim.x;
  for (long long i = (long long)blockIdx.x * blockDim.x + threadIdx.x; i < group_size; i += stride) {
    float v = xg[i]; s += v; s2 += v * v;
  }
  s  = block_sum(s, sb);
  s2 = block_sum(s2, sb);
  if (threadIdx.x == 0) {
    atomicAdd(&stats[g * 2 + 0], s);
    atomicAdd(&stats[g * 2 + 1], s2);
  }
}

// instance-norm (mean/var from stats) then row softmax, in place; rows contiguous [row][cols]
__global__ void inorm_softmax_kernel(float* __restrict__ x, int cols,
                                     const float* __restrict__ stats,
                                     int rows_per_group, float inv_count, float eps)
{
  __shared__ float sb[256];
  int row = blockIdx.x;
  int g = row / rows_per_group;
  float mean = stats[g * 2 + 0] * inv_count;
  float var  = stats[g * 2 + 1] * inv_count - mean * mean;
  float rstd = rsqrtf(var + eps);
  float* xr = x + (long long)row * cols;

  float mx = -3.0e38f;
  for (int c = threadIdx.x; c < cols; c += blockDim.x)
    mx = fmaxf(mx, (xr[c] - mean) * rstd);
  mx = block_max(mx, sb);

  float s = 0.f;
  for (int c = threadIdx.x; c < cols; c += blockDim.x) {
    float e = expf((xr[c] - mean) * rstd - mx);
    xr[c] = e; s += e;
  }
  s = block_sum(s, sb);
  float inv = 1.f / s;
  for (int c = threadIdx.x; c < cols; c += blockDim.x) xr[c] *= inv;
}

// fused decoder bias: ball[o] = sum_blocks fuse_w[o, blk] . c{4,3,2,1}_b
__global__ void fuse_bias_kernel(const float* __restrict__ fw,
                                 const float* __restrict__ b4, const float* __restrict__ b3,
                                 const float* __restrict__ b2, const float* __restrict__ b1,
                                 float* __restrict__ ball)
{
  int o = blockIdx.x * blockDim.x + threadIdx.x;
  if (o >= 768) return;
  const float* r = fw + (long long)o * 3072;
  float s = 0.f;
  for (int j = 0; j < 768; ++j) s += r[j]        * b4[j];
  for (int j = 0; j < 768; ++j) s += r[768 + j]  * b3[j];
  for (int j = 0; j < 768; ++j) s += r[1536 + j] * b2[j];
  for (int j = 0; j < 768; ++j) s += r[2304 + j] * b1[j];
  ball[o] = s;
}

// U[b][p][k] (bf16), p over 128x128, k: [0,512)=up(e4n,16), [512,832)=up(e3,32),
// [832,960)=up(e2,64), [960,1024)=e1 native
__global__ void build_U_kernel(const float* __restrict__ e4n, const float* __restrict__ e3,
                               const float* __restrict__ e2, const float* __restrict__ e1,
                               bf16_t* __restrict__ U, long long total)
{
  long long idx = (long long)blockIdx.x * blockDim.x + threadIdx.x;
  if (idx >= total) return;
  int k = (int)(idx & 1023);
  long long rem = idx >> 10;
  int p = (int)(rem & 16383);
  int b = (int)(rem >> 14);
  int py = p >> 7, px = p & 127;
  float v;
  if (k < 512) {
    const float* pl = e4n + ((long long)(b * 512 + k)) * 256;
    float sc = 16.f / 128.f;
    v = sample_bilinear(pl, 16, (py + 0.5f) * sc - 0.5f, (px + 0.5f) * sc - 0.5f);
  } else if (k < 832) {
    const float* pl = e3 + ((long long)(b * 320 + (k - 512))) * 1024;
    float sc = 32.f / 128.f;
    v = sample_bilinear(pl, 32, (py + 0.5f) * sc - 0.5f, (px + 0.5f) * sc - 0.5f);
  } else if (k < 960) {
    const float* pl = e2 + ((long long)(b * 128 + (k - 832))) * 4096;
    float sc = 64.f / 128.f;
    v = sample_bilinear(pl, 64, (py + 0.5f) * sc - 0.5f, (px + 0.5f) * sc - 0.5f);
  } else {
    v = e1[(((long long)(b * 64 + (k - 960))) << 14) + p];
  }
  U[idx] = (bf16_t)v;
}

// per-channel mean / rstd over (B, spatial); y is [b][p][o] bf16, 32768 rows
__global__ void chan_stats_kernel(const bf16_t* __restrict__ y, float* __restrict__ st) {
  __shared__ float sb[256];
  int o = blockIdx.x;
  float s = 0.f, s2 = 0.f;
  for (int i = threadIdx.x; i < 32768; i += blockDim.x) {
    float v = (float)y[(long long)i * 768 + o];
    s += v; s2 += v * v;
  }
  s  = block_sum(s, sb);
  s2 = block_sum(s2, sb);
  if (threadIdx.x == 0) {
    float mean = s / 32768.f;
    float var  = s2 / 32768.f - mean * mean;
    st[o]       = mean;
    st[768 + o] = rsqrtf(var + 1e-5f);
  }
}

// in-place: y = relu((y-mean)*rstd*g + b)
__global__ void norm_relu_kernel(bf16_t* __restrict__ y, const float* __restrict__ st,
                                 const float* __restrict__ g, const float* __restrict__ b,
                                 long long total)
{
  long long idx = (long long)blockIdx.x * blockDim.x + threadIdx.x;
  if (idx >= total) return;
  int o = (int)(idx % 768);
  float v = ((float)y[idx] - st[o]) * st[768 + o] * g[o] + b[o];
  y[idx] = (bf16_t)fmaxf(v, 0.f);
}

// bilinear 128x128 -> 512x512 for 2*21 planes into d_out
__global__ void resize_out_kernel(const float* __restrict__ logits, float* __restrict__ out,
                                  long long total)
{
  long long idx = (long long)blockIdx.x * blockDim.x + threadIdx.x;
  if (idx >= total) return;
  long long perb = 21LL * 262144;
  int b  = (int)(idx / perb);
  long long rem = idx - (long long)b * perb;
  int j  = (int)(rem / 262144);
  int pr = (int)(rem % 262144);
  int Y = pr >> 9, X = pr & 511;
  const float* pl = logits + ((long long)(b * 21 + j)) * 16384;
  float sc = 128.f / 512.f;
  out[idx] = sample_bilinear(pl, 128, (Y + 0.5f) * sc - 0.5f, (X + 0.5f) * sc - 0.5f);
}

// ======================= host-side launcher =======================
static inline void launch_gemm(hipStream_t st,
    const void* A, int a_bf, long long a_rs, long long a_cs, long long a_bs,
    const void* Bm, int b_bf, long long b_rs, long long b_cs, long long b_bs,
    void* Cm, int c_bf, long long c_rs, long long c_cs, long long c_bs,
    const float* bias, int act, int M, int N, int K, int Z)
{
  dim3 grid((N + 63) / 64, (M + 63) / 64, Z);
  dim3 block(32, 4, 1);
  bool guard = (M & 63) || (N & 63) || (K & 31);
#define LG(AB, BB, CB, GD)                                                          \
  gemm_wmma_t<AB, BB, CB, GD><<<grid, block, 0, st>>>(A, a_rs, a_cs, a_bs,          \
                                                      Bm, b_rs, b_cs, b_bs,         \
                                                      Cm, c_rs, c_cs, c_bs,         \
                                                      bias, act, M, N, K)
#define LG2(AB, BB, CB) do { if (guard) LG(AB, BB, CB, 1); else LG(AB, BB, CB, 0); } while (0)
  if      (!a_bf && !b_bf && !c_bf) LG2(0, 0, 0);
  else if (!a_bf && !b_bf &&  c_bf) LG2(0, 0, 1);
  else if ( a_bf &&  b_bf &&  c_bf) LG2(1, 1, 1);
  else                              LG2(1, 0, 0);
#undef LG2
#undef LG
}

extern "C" void kernel_launch(void* const* d_in, const int* in_sizes, int n_in,
                              void* d_out, int out_size, void* d_ws, size_t ws_size,
                              hipStream_t stream)
{
  (void)in_sizes; (void)n_in; (void)out_size; (void)ws_size;

  const float* e1    = (const float*)d_in[0];
  const float* e2    = (const float*)d_in[1];
  const float* e3    = (const float*)d_in[2];
  const float* e4    = (const float*)d_in[3];
  // d_in[4], d_in[5] are h=512, w=512 (device scalars; output size fixed by out_size => hardcoded)
  const float* mi_w  = (const float*)d_in[6];
  const float* mi_b  = (const float*)d_in[7];
  const float* an_g  = (const float*)d_in[8];
  const float* an_b  = (const float*)d_in[9];
  const float* wq_uu = (const float*)d_in[10];
  const float* wk_uu = (const float*)d_in[11];
  const float* wv_uu = (const float*)d_in[12];
  const float* wq_lu = (const float*)d_in[13];
  const float* wk_lu = (const float*)d_in[14];
  const float* wv_lu = (const float*)d_in[15];
  const float* wo_u  = (const float*)d_in[16];
  const float* wo_l  = (const float*)d_in[17];
  const float* en_g  = (const float*)d_in[18];
  const float* en_b  = (const float*)d_in[19];
  const float* mo_w  = (const float*)d_in[20];
  const float* mo_b  = (const float*)d_in[21];
  const float* c4_w  = (const float*)d_in[22];
  const float* c4_b  = (const float*)d_in[23];
  const float* c3_w  = (const float*)d_in[24];
  const float* c3_b  = (const float*)d_in[25];
  const float* c2_w  = (const float*)d_in[26];
  const float* c2_b  = (const float*)d_in[27];
  const float* c1_w  = (const float*)d_in[28];
  const float* c1_b  = (const float*)d_in[29];
  const float* fuse_w = (const float*)d_in[30];
  const float* fuse_g = (const float*)d_in[31];
  const float* fuse_b = (const float*)d_in[32];
  const float* pred_w = (const float*)d_in[33];
  const float* pred_b = (const float*)d_in[34];

  float* out_seg = (float*)d_out;                    // (2,21,512,512)
  float* e4n_out = out_seg + 2LL * 21 * 512 * 512;   // (2,512,16,16)

  // ---- workspace layout (stage 1: messenger; ~42 MB) ----
  char* W = (char*)d_ws;
  float* t_en   = (float*)W;               // (2,256,512)
  float* t_emb  = t_en   + 262144;         // (2,256,512)
  float* Qu     = t_emb  + 262144;         // (256,2048) x6
  float* Ku     = Qu     + 524288;
  float* Vu     = Ku     + 524288;
  float* Ql     = Vu     + 524288;
  float* Kf     = Ql     + 524288;
  float* Vf     = Kf     + 524288;
  float* attn4  = Vf     + 524288;         // (4,512,512)
  float* attn2  = attn4  + 1048576;        // (2048,2048)
  float* ctxb   = attn2  + 4194304;        // (256,2048) interleaved head layout
  float* ctx2   = ctxb   + 524288;         // (256,2048)
  float* t_O    = ctx2   + 524288;         // (2,256,512)
  float* t_out  = t_O    + 262144;         // (2,256,512)
  float* mstats = t_out  + 262144;         // 32 floats: [0..7] per-head, [8..9] cross

  // ---- stage 2 buffers reuse the same region (messenger finished first) ----
  bf16_t* Wcat  = (bf16_t*)W;                              // (1024,768) bf16 fused weights
  float*  ball  = (float*)(W + (2LL << 20));               // 768
  float*  cst   = ball + 1024;                             // 1536 (mean, rstd)
  float*  logits = cst + 2048;                             // (2,21,128,128)
  bf16_t* U     = (bf16_t*)(W + (8LL << 20));              // (2,16384,1024) bf16, 64 MB
  bf16_t* yb    = (bf16_t*)(W + (8LL << 20) + 67108864LL); // (2,16384,768) bf16, 48 MB
  // total scratch required ~126 MB

  // ================== stage 1: SD messenger on e4 ==================
  zero_kernel<<<1, 32, 0, stream>>>(mstats, 32);

  // en[b][n][c] = gelu(sum_k mi_w[c][k] * e4[b][k][n] + mi_b[c])
  launch_gemm(stream, e4, 0, 1, 256, 131072,  mi_w, 0, 1, 512, 0,
              t_en, 0, 512, 1, 131072,  mi_b, 1, 256, 512, 512, 2);
  // emb = LN(en)
  ln_rows_kernel<<<512, 256, 0, stream>>>(t_en, nullptr, an_g, an_b, t_emb, 512, 1e-6f);

  const float* emb_l = t_emb;            // batch 0
  const float* emb_u = t_emb + 131072;   // batch 1

  // projections (256,2048) = emb @ W.T   (K=512)
  launch_gemm(stream, emb_u, 0, 512, 1, 0,  wq_uu, 0, 1, 512, 0,  Qu, 0, 2048, 1, 0, nullptr, 0, 256, 2048, 512, 1);
  launch_gemm(stream, emb_u, 0, 512, 1, 0,  wk_uu, 0, 1, 512, 0,  Ku, 0, 2048, 1, 0, nullptr, 0, 256, 2048, 512, 1);
  launch_gemm(stream, emb_u, 0, 512, 1, 0,  wv_uu, 0, 1, 512, 0,  Vu, 0, 2048, 1, 0, nullptr, 0, 256, 2048, 512, 1);
  launch_gemm(stream, emb_l, 0, 512, 1, 0,  wq_lu, 0, 1, 512, 0,  Ql, 0, 2048, 1, 0, nullptr, 0, 256, 2048, 512, 1);
  launch_gemm(stream, emb_u, 0, 512, 1, 0,  wk_lu, 0, 1, 512, 0,  Kf, 0, 2048, 1, 0, nullptr, 0, 256, 2048, 512, 1);
  launch_gemm(stream, emb_u, 0, 512, 1, 0,  wv_lu, 0, 1, 512, 0,  Vf, 0, 2048, 1, 0, nullptr, 0, 256, 2048, 512, 1);

  // attn[h][c][d] = sum_n Qu[n, h*512+c] * Ku[n, h*512+d]
  launch_gemm(stream, Qu, 0, 1, 2048, 512,  Ku, 0, 2048, 1, 512,
              attn4, 0, 512, 1, 262144, nullptr, 0, 512, 512, 256, 4);
  reduce_sums_kernel<<<dim3(64, 1, 4), 256, 0, stream>>>(attn4, 262144LL, mstats);
  inorm_softmax_kernel<<<4 * 512, 256, 0, stream>>>(attn4, 512, mstats, 512, 1.f / 262144.f, 1e-5f);

  // ctx[n][c*4+h] = sum_d sim[h][c][d] * Vu[n][h*512+d]
  launch_gemm(stream, attn4, 0, 512, 1, 262144,  Vu, 0, 1, 2048, 512,
              ctxb, 0, 4, 2048, 1, nullptr, 0, 512, 256, 512, 4);
  // O_u = ctx @ wo_u.T  -> t_O batch 1
  launch_gemm(stream, ctxb, 0, 2048, 1, 0,  wo_u, 0, 1, 2048, 0,
              t_O + 131072, 0, 512, 1, 0, nullptr, 0, 256, 512, 2048, 1);

  // attn2[c][d] = sum_n Ql[n][c] * Kf[n][d]  (2048x2048)
  launch_gemm(stream, Ql, 0, 1, 2048, 0,  Kf, 0, 2048, 1, 0,
              attn2, 0, 2048, 1, 0, nullptr, 0, 2048, 2048, 256, 1);
  reduce_sums_kernel<<<dim3(256, 1, 1), 256, 0, stream>>>(attn2, 4194304LL, mstats + 8);
  inorm_softmax_kernel<<<2048, 256, 0, stream>>>(attn2, 2048, mstats + 8, 2048, 1.f / 4194304.f, 1e-5f);

  // ctx2[n][c] = sum_d sim2[c][d] * Vf[n][d]
  launch_gemm(stream, Vf, 0, 2048, 1, 0,  attn2, 0, 1, 2048, 0,
              ctx2, 0, 2048, 1, 0, nullptr, 0, 256, 2048, 2048, 1);
  // O_l = ctx2 @ wo_l.T -> t_O batch 0
  launch_gemm(stream, ctx2, 0, 2048, 1, 0,  wo_l, 0, 1, 2048, 0,
              t_O, 0, 512, 1, 0, nullptr, 0, 256, 512, 2048, 1);

  // emb2 = O + en (residual), LN(en_g, en_b)
  ln_rows_kernel<<<512, 256, 0, stream>>>(t_O, t_en, en_g, en_b, t_out, 512, 1e-6f);

  // e4n[b][o][p] = gelu(sum_c mo_w[o][c] * t_out[b][p][c] + mo_b[o])  -> d_out tail
  launch_gemm(stream, t_out, 0, 512, 1, 131072,  mo_w, 0, 1, 512, 0,
              e4n_out, 0, 1, 256, 131072,  mo_b, 1, 256, 512, 512, 2);

  // ================== stage 2: decoder (resize/concat/conv folded) ==================
  // Wcat[k][o] bf16: fused weights  W_i = fuse_w[:, blk_i] @ c_i_w
  launch_gemm(stream, fuse_w,        0, 3072, 1, 0,  c4_w, 0, 512, 1, 0,
              Wcat,             1, 1, 768, 0, nullptr, 0, 768, 512, 768, 1);
  launch_gemm(stream, fuse_w + 768,  0, 3072, 1, 0,  c3_w, 0, 320, 1, 0,
              Wcat + 512 * 768, 1, 1, 768, 0, nullptr, 0, 768, 320, 768, 1);
  launch_gemm(stream, fuse_w + 1536, 0, 3072, 1, 0,  c2_w, 0, 128, 1, 0,
              Wcat + 832 * 768, 1, 1, 768, 0, nullptr, 0, 768, 128, 768, 1);
  launch_gemm(stream, fuse_w + 2304, 0, 3072, 1, 0,  c1_w, 0, 64, 1, 0,
              Wcat + 960 * 768, 1, 1, 768, 0, nullptr, 0, 768, 64, 768, 1);
  fuse_bias_kernel<<<3, 256, 0, stream>>>(fuse_w, c4_b, c3_b, c2_b, c1_b, ball);

  // U[b][p][k]: bilinear-interpolated inputs at 128x128, bf16
  long long totU = 2LL * 16384 * 1024;
  build_U_kernel<<<(unsigned)((totU + 255) / 256), 256, 0, stream>>>(e4n_out, e3, e2, e1, U, totU);

  // y[b][p][o] = U @ Wcat + ball   (M=16384, N=768, K=1024, Z=2) -> bf16
  launch_gemm(stream, U, 1, 1024, 1, 16777216LL,  Wcat, 1, 768, 1, 0,
              yb, 1, 768, 1, 12582912LL,  ball, 0, 16384, 768, 1024, 2);

  // per-channel norm over (B, spatial) + affine + ReLU (in place)
  chan_stats_kernel<<<768, 256, 0, stream>>>(yb, cst);
  long long totY = 2LL * 16384 * 768;
  norm_relu_kernel<<<(unsigned)((totY + 255) / 256), 256, 0, stream>>>(yb, cst, fuse_g, fuse_b, totY);

  // logits[b][j][p] = y @ pred_w.T + pred_b  (N=21)
  launch_gemm(stream, yb, 1, 768, 1, 12582912LL,  pred_w, 0, 1, 768, 0,
              logits, 0, 1, 16384, 344064LL,  pred_b, 0, 16384, 21, 768, 2);

  // final bilinear 128->512 into d_out
  long long totO = 2LL * 21 * 512 * 512;
  resize_out_kernel<<<(unsigned)((totO + 255) / 256), 256, 0, stream>>>(logits, out_seg, totO);
}
